// FactorRGCN_23656679866462
// MI455X (gfx1250) — matrix operand order repaired and checked by reading, not compiled, not run on hardware
//
#include <hip/hip_runtime.h>
#include <stdint.h>

#define N_NODES 50000
#define N_EDGES 1000000
#define IN_DIM  64
#define HID_DIM 128
#define OUT_DIM 64
#define N_REL   8

typedef __attribute__((ext_vector_type(16))) _Float16 v16h;
typedef __attribute__((ext_vector_type(8)))  _Float16 v8h;
typedef __attribute__((ext_vector_type(8)))  float    v8f;

#define LDS_STRIDE 40  // halfs; 80B row stride -> 20-bank stride, conflict-free b128 loads

// ---------------- utility kernels ----------------

__global__ void zero_f32(float* __restrict__ p, int n) {
  int i = blockIdx.x * blockDim.x + threadIdx.x;
  int stride = gridDim.x * blockDim.x;
  for (; i < n; i += stride) p[i] = 0.0f;
}

__global__ void gather_emb(const int* __restrict__ x, const float* __restrict__ emb,
                           float* __restrict__ h0) {
  int i = blockIdx.x * blockDim.x + threadIdx.x;
  if (i >= N_NODES * IN_DIM) return;
  int n = i >> 6, f = i & 63;
  h0[i] = emb[(long long)x[n] * IN_DIM + f];
}

__global__ void edge_count(const int* __restrict__ ei, const int* __restrict__ et,
                           float* __restrict__ cnt) {
  int e = blockIdx.x * blockDim.x + threadIdx.x;
  if (e >= N_EDGES) return;
  int dst = ei[N_EDGES + e];
  atomicAdd(&cnt[dst * N_REL + et[e]], 1.0f);
}

__global__ void inv_cnt(float* __restrict__ cnt) {
  int i = blockIdx.x * blockDim.x + threadIdx.x;
  if (i >= N_NODES * N_REL) return;
  float c = cnt[i];
  cnt[i] = (c > 0.0f) ? (1.0f / c) : 0.0f;
}

// scatter-sum of source features into acc[(dst*8+etype)*FIN + f]
template <int FIN>
__global__ void edge_scatter(const int* __restrict__ ei, const int* __restrict__ et,
                             const float* __restrict__ H, float* __restrict__ acc) {
  const int CH = FIN / 4;  // float4 chunks per edge
  long long tid = (long long)blockIdx.x * blockDim.x + threadIdx.x;
  if (tid >= (long long)N_EDGES * CH) return;
  int e = (int)(tid / CH);
  int c = (int)(tid % CH);
  int src = ei[e];
  int dst = ei[N_EDGES + e];
  int r   = et[e];
  const float4 v = *(const float4*)(H + (long long)src * FIN + c * 4);
  float* a = acc + ((long long)dst * N_REL + r) * FIN + c * 4;
  atomicAdd(a + 0, v.x);
  atomicAdd(a + 1, v.y);
  atomicAdd(a + 2, v.z);
  atomicAdd(a + 3, v.w);
}

// Transposed stacked weights: WstkT[n * KTOT + k], k rows 0..8*FIN-1 = W_r, 8*FIN.. = W_root
template <int FIN, int NOUT>
__global__ void build_wstkT(const float* __restrict__ W, const float* __restrict__ Wroot,
                            _Float16* __restrict__ WstkT) {
  const int KTOT = (N_REL + 1) * FIN;
  const int TOT  = KTOT * NOUT;
  int i = blockIdx.x * blockDim.x + threadIdx.x;
  if (i >= TOT) return;
  int n = i / KTOT, k = i % KTOT;
  int r = k / FIN, f = k % FIN;
  float v = (r < N_REL) ? W[((long long)r * FIN + f) * NOUT + n]
                        : Wroot[(long long)f * NOUT + n];
  WstkT[i] = (_Float16)v;
}

// ---------------- WMMA GEMM ----------------
// C[16, NOUT] per block; A row = [mean(acc, r=0..7) | Hin], width 9*FIN,
// invcnt-normalized + f32->f16 converted while staging into LDS.
// One wave per 16-col tile (NT waves). K-loop in steps of 32 with
// v_wmma_f32_16x16x32_f16; fragments loaded as contiguous 16B LDS vectors.
template <int NT, int FSHIFT, int NOUT, bool RELU>
__global__ __launch_bounds__(NT * 32) void rgcn_gemm(
    const float* __restrict__ Hin, const float* __restrict__ acc,
    const float* __restrict__ invc, const _Float16* __restrict__ WstkT,
    const float* __restrict__ bias, float* __restrict__ out) {
  const int FIN   = 1 << FSHIFT;
  const int KTOT  = (N_REL + 1) * FIN;
  const int KSTEP = KTOT / 32;

  __shared__ alignas(16) _Float16 ldsA[16 * LDS_STRIDE];    // [row][k], padded
  __shared__ alignas(16) _Float16 ldsB[NOUT * LDS_STRIDE];  // [col][k], padded

  const int tid  = threadIdx.x;
  const int lane = tid & 31;
  const int wave = tid >> 5;     // column-tile index
  const int hi   = lane >> 4;    // 0: lanes 0-15, 1: lanes 16-31
  const int m    = lane & 15;
  const int rowBase = blockIdx.x * 16;
  const int col     = wave * 16 + m;

  v8f c = {};

  for (int kb = 0; kb < KSTEP; ++kb) {
    // Stage A tile (16 rows x 32 k): fused mean-normalize + f16 convert.
    for (int idx = tid; idx < 16 * 32; idx += NT * 32) {
      int row = idx >> 5, kk = idx & 31;
      int gRow = rowBase + row;
      int k = kb * 32 + kk;
      int r = k >> FSHIFT;
      int f = k & (FIN - 1);
      float v;
      if (r < N_REL)
        v = acc[((long long)gRow * N_REL + r) * FIN + f] * invc[gRow * N_REL + r];
      else
        v = Hin[(long long)gRow * FIN + f];
      ldsA[row * LDS_STRIDE + kk] = (_Float16)v;
    }
    // Stage B tile from transposed weights: 32 k-halfs per column, u32 copies.
    for (int idx = tid; idx < 16 * NOUT; idx += NT * 32) {
      int n = idx >> 4, u = idx & 15;  // u: pair-of-halfs index within the 32-k run
      ((uint32_t*)ldsB)[n * (LDS_STRIDE / 2) + u] =
          ((const uint32_t*)WstkT)[(long long)n * (KTOT / 2) + kb * 16 + u];
    }
    __syncthreads();

    // A fragment: lane m holds K = hi*8 + (0..7) and 16 + hi*8 + (0..7)
    // (CDNA5 16-bit A 16x32 layout) -> two contiguous 16B LDS loads.
    v8h alo = *(const v8h*)(ldsA + m * LDS_STRIDE + hi * 8);
    v8h ahi = *(const v8h*)(ldsA + m * LDS_STRIDE + 16 + hi * 8);
    v16h a = __builtin_shufflevector(alo, ahi, 0, 1, 2, 3, 4, 5, 6, 7, 8, 9, 10,
                                     11, 12, 13, 14, 15);
    // B fragment: lane holds column `col`, K = hi*8 + (0..7) and 16 + hi*8 + (0..7).
    v8h blo = *(const v8h*)(ldsB + col * LDS_STRIDE + hi * 8);
    v8h bhi = *(const v8h*)(ldsB + col * LDS_STRIDE + 16 + hi * 8);
    v16h b = __builtin_shufflevector(blo, bhi, 0, 1, 2, 3, 4, 5, 6, 7, 8, 9, 10,
                                     11, 12, 13, 14, 15);

    c = __builtin_amdgcn_wmma_f32_16x16x32_f16(
        /*neg_a=*/false, a, /*neg_b=*/false, b,
        /*c_mod=*/(short)0, c, /*reuse_a=*/false, /*reuse_b=*/false);
    __syncthreads();
  }

  // Epilogue: C/D layout -> VGPR i holds row (i + 8*hi), lane m is column.
  const float bv = bias[col];
#pragma unroll
  for (int i = 0; i < 8; ++i) {
    int row = rowBase + i + (hi ? 8 : 0);
    float v = c[i] + bv;
    if (RELU) v = fmaxf(v, 0.0f);
    out[(long long)row * NOUT + col] = v;
  }
}

// ---------------- launch ----------------

static inline int cdiv_ll(long long a, long long b) { return (int)((a + b - 1) / b); }

extern "C" void kernel_launch(void* const* d_in, const int* in_sizes, int n_in,
                              void* d_out, int out_size, void* d_ws, size_t ws_size,
                              hipStream_t stream) {
  const int*   x   = (const int*)d_in[0];
  const int*   ei  = (const int*)d_in[1];   // [2, E]
  const int*   et  = (const int*)d_in[2];   // [E]
  const float* emb = (const float*)d_in[3]; // [N, 64]
  const float* W1  = (const float*)d_in[4]; // [8, 64, 128]
  const float* W1r = (const float*)d_in[5]; // [64, 128]
  const float* b1  = (const float*)d_in[6]; // [128]
  const float* W2  = (const float*)d_in[7]; // [8, 128, 64]
  const float* W2r = (const float*)d_in[8]; // [128, 64]
  const float* b2  = (const float*)d_in[9]; // [64]
  float* out = (float*)d_out;               // [N, 64]

  // workspace layout (~245 MB)
  float* cnt = (float*)d_ws;                              // N*R
  float* acc = cnt + (size_t)N_NODES * N_REL;             // N*R*HID (max of layers)
  float* h0  = acc + (size_t)N_NODES * N_REL * HID_DIM;   // N*IN
  float* h1  = h0  + (size_t)N_NODES * IN_DIM;            // N*HID
  _Float16* wstk = (_Float16*)(h1 + (size_t)N_NODES * HID_DIM);  // 9*128*128 max

  // edge counts -> reciprocals (identical for both layers)
  zero_f32<<<cdiv_ll(N_NODES * N_REL, 256), 256, 0, stream>>>(cnt, N_NODES * N_REL);
  edge_count<<<cdiv_ll(N_EDGES, 256), 256, 0, stream>>>(ei, et, cnt);
  inv_cnt<<<cdiv_ll(N_NODES * N_REL, 256), 256, 0, stream>>>(cnt);

  // h0 = node_emb[x]
  gather_emb<<<cdiv_ll((long long)N_NODES * IN_DIM, 256), 256, 0, stream>>>(x, emb, h0);

  // ---- Layer 1: [N,576]x[576,128] + bias, ReLU ----
  zero_f32<<<cdiv_ll((long long)N_NODES * N_REL * IN_DIM, 256), 256, 0, stream>>>(
      acc, N_NODES * N_REL * IN_DIM);
  edge_scatter<IN_DIM><<<cdiv_ll((long long)N_EDGES * (IN_DIM / 4), 256), 256, 0, stream>>>(
      ei, et, h0, acc);
  build_wstkT<IN_DIM, HID_DIM><<<cdiv_ll((N_REL + 1) * IN_DIM * HID_DIM, 256), 256, 0, stream>>>(
      W1, W1r, wstk);
  rgcn_gemm<8, 6, HID_DIM, true><<<N_NODES / 16, 8 * 32, 0, stream>>>(
      h0, acc, cnt, wstk, b1, h1);

  // ---- Layer 2: [N,1152]x[1152,64] + bias ----
  zero_f32<<<cdiv_ll((long long)N_NODES * N_REL * HID_DIM, 256), 256, 0, stream>>>(
      acc, N_NODES * N_REL * HID_DIM);
  edge_scatter<HID_DIM><<<cdiv_ll((long long)N_EDGES * (HID_DIM / 4), 256), 256, 0, stream>>>(
      ei, et, h1, acc);
  build_wstkT<HID_DIM, OUT_DIM><<<cdiv_ll((N_REL + 1) * HID_DIM * OUT_DIM, 256), 256, 0, stream>>>(
      W2, W2r, wstk);
  rgcn_gemm<4, 7, OUT_DIM, false><<<N_NODES / 16, 4 * 32, 0, stream>>>(
      h1, acc, cnt, wstk, b2, out);
}
